// LinearAttention_45552423141626
// MI455X (gfx1250) — matrix operand order repaired and checked by reading, not compiled, not run on hardware
//
#include <hip/hip_runtime.h>

// ---------------------------------------------------------------------------
// Linear attention for MI455X (gfx1250), wave32 + v_wmma_f32_16x16x32_bf16.
// All contractions phrased as out[n][m] = sum_k A[m][k]*B[n][k] (NT form) so
// both operand fragment loads are contiguous b128 per the CDNA5 WMMA VGPR
// layouts, and the transposed store OUT[n][m] is per-lane contiguous.
// Big GEMMs: 4-wave workgroups on 128x128 block tiles staged through LDS
// (double-buffered, async global->LDS DMA when the builtin exists) to halve
// L2 traffic vs independent 64x64 wave tiles.
// ---------------------------------------------------------------------------

typedef __attribute__((ext_vector_type(16))) __bf16 v16bf;
typedef __attribute__((ext_vector_type(8)))  __bf16 v8bf;
typedef __attribute__((ext_vector_type(8)))  float  v8f;
typedef __attribute__((ext_vector_type(4)))  float  v4f;
typedef __attribute__((ext_vector_type(4)))  int    v4i;

#define BATCH   4
#define SEQ     4096
#define TOKENS  16384   // BATCH * SEQ
#define DIM     1024
#define HEADS   16
#define HDIM    64
#define KVPARTS 16
#define EPSV    1e-6f
#define LDSROW  40      // 32 bf16 + 8 pad: 80B row stride -> conflict-free b128 reads

// Epilogue flags
#define FLAG_ELU    1
#define FLAG_BIAS_M 2
#define FLAG_BIAS_N 4
#define FLAG_NORM   8
#define FLAG_OUTF32 16

// -------------------- async global->LDS (CDNA5 DMA path) -------------------
#if defined(__HIP_DEVICE_COMPILE__) && __has_builtin(__builtin_amdgcn_global_load_async_to_lds_b128)
#define HAVE_ASYNC_LDS 1
#else
#define HAVE_ASYNC_LDS 0
#endif

#if HAVE_ASYNC_LDS
__device__ __forceinline__ void async_cp16(const __bf16* g, __bf16* l) {
    __builtin_amdgcn_global_load_async_to_lds_b128((v4i*)g, (v4i*)l, 0, 0);
}
#endif

template<int N>
__device__ __forceinline__ void wait_async() {
#if HAVE_ASYNC_LDS
#if __has_builtin(__builtin_amdgcn_s_wait_asynccnt)
    __builtin_amdgcn_s_wait_asynccnt(N);
#else
    asm volatile("s_wait_asynccnt %0" :: "n"(N));
#endif
#endif
}

// ---------------------- shared transposed-store epilogue -------------------
template<int FLAGS>
__device__ __forceinline__ void epi_store(void* __restrict__ OUT, int ldo,
                                          int m0, int n0,
                                          const v8f (&acc)[4][4], int lane,
                                          const float* __restrict__ bias_m,
                                          const float* __restrict__ bias_n,
                                          const float* __restrict__ norm_n)
{
    const int g = lane >> 4, lr = lane & 15;
#pragma unroll
    for (int ni = 0; ni < 4; ++ni) {
        const int n = n0 + 16 * ni + lr;
        float addn = 0.0f, rnorm = 1.0f;
        if (FLAGS & FLAG_BIAS_N) addn = bias_n[n];
        if (FLAGS & FLAG_NORM)   rnorm = 1.0f / norm_n[n];
#pragma unroll
        for (int mi = 0; mi < 4; ++mi) {
            const int m = m0 + 16 * mi + 8 * g;
            float vals[8];
#pragma unroll
            for (int r = 0; r < 8; ++r) {
                float v = acc[mi][ni][r];
                if (FLAGS & FLAG_BIAS_N) v += addn;
                if (FLAGS & FLAG_BIAS_M) v += bias_m[m + r];
                if (FLAGS & FLAG_ELU)    v = (v > 0.0f) ? (v + 1.0f) : __expf(v);
                if (FLAGS & FLAG_NORM)   v *= rnorm;
                vals[r] = v;
            }
            if (FLAGS & FLAG_OUTF32) {
                float* o = (float*)OUT + (size_t)n * ldo + m;
                v4f a, b;
#pragma unroll
                for (int r = 0; r < 4; ++r) { a[r] = vals[r]; b[r] = vals[4 + r]; }
                *(v4f*)(o) = a;
                *(v4f*)(o + 4) = b;
            } else {
                v8bf ob;
#pragma unroll
                for (int r = 0; r < 8; ++r) ob[r] = (__bf16)vals[r];
                *(v8bf*)((__bf16*)OUT + (size_t)n * ldo + m) = ob;
            }
        }
    }
}

// ------------- register-resident 64x64 wave GEMM (small stages) ------------
template<int FLAGS>
__device__ __forceinline__ void wave_gemm_64x64(
    const __bf16* __restrict__ A, int lda,
    const __bf16* __restrict__ B, int ldb,
    void* __restrict__ OUT, int ldo,
    int m0, int n0, int K,
    const float* __restrict__ bias_m,
    const float* __restrict__ bias_n,
    const float* __restrict__ norm_n)
{
    const int lane = threadIdx.x & 31;
    const int g = lane >> 4, lr = lane & 15;
    v8f acc[4][4] = {};

    for (int k0 = 0; k0 < K; k0 += 32) {
        v16bf af[4], bfr[4];
#pragma unroll
        for (int mi = 0; mi < 4; ++mi) {
            const __bf16* p = A + (size_t)(m0 + 16 * mi + lr) * lda + k0 + 8 * g;
            v8bf lo = *(const v8bf*)(p);
            v8bf hi = *(const v8bf*)(p + 16);
#pragma unroll
            for (int e = 0; e < 8; ++e) { af[mi][e] = lo[e]; af[mi][8 + e] = hi[e]; }
        }
#pragma unroll
        for (int ni = 0; ni < 4; ++ni) {
            const __bf16* p = B + (size_t)(n0 + 16 * ni + lr) * ldb + k0 + 8 * g;
            v8bf lo = *(const v8bf*)(p);
            v8bf hi = *(const v8bf*)(p + 16);
#pragma unroll
            for (int e = 0; e < 8; ++e) { bfr[ni][e] = lo[e]; bfr[ni][8 + e] = hi[e]; }
        }
#pragma unroll
        for (int mi = 0; mi < 4; ++mi)
#pragma unroll
            for (int ni = 0; ni < 4; ++ni)
                acc[mi][ni] = __builtin_amdgcn_wmma_f32_16x16x32_bf16(
                    false, af[mi], false, bfr[ni], (short)0, acc[mi][ni], false, false);
    }
    epi_store<FLAGS>(OUT, ldo, m0, n0, acc, lane, bias_m, bias_n, norm_n);
}

// ------------- LDS-staged 128x128 block GEMM (4 waves, big stages) ---------
template<int FLAGS>
__global__ __launch_bounds__(128)
void k_gemm_lds(const __bf16* __restrict__ A, int lda,
                const __bf16* __restrict__ B, int ldb,
                void* __restrict__ OUT, int ldo, int K,
                const float* __restrict__ bias_m,
                const float* __restrict__ bias_n)
{
    __shared__ __bf16 sA[2][128 * LDSROW];   // 10 KB per buffer
    __shared__ __bf16 sB[2][128 * LDSROW];   // 40 KB total (double-buffered)

    const int tid  = threadIdx.x;
    const int lane = tid & 31;
    const int wave = tid >> 5;                      // 0..3
    const int wm = (wave & 1) * 64, wn = (wave >> 1) * 64;
    const int g = lane >> 4, lr = lane & 15;
    const int bm = blockIdx.x * 128, bn = blockIdx.y * 128;

    // Stage one K=32 slice of both operands: 512 b128 transfers per matrix,
    // 4 per thread per matrix -> 8 ASYNCcnt increments per wave per stage.
    auto issue = [&](int k0, int buf) {
#pragma unroll
        for (int j = 0; j < 4; ++j) {
            const int f = j * 128 + tid;            // 0..511
            const int row = f >> 2, c16 = f & 3;    // 4 x 16B per 64B row
            const __bf16* ga = A + (size_t)(bm + row) * lda + k0 + c16 * 8;
            const __bf16* gb = B + (size_t)(bn + row) * ldb + k0 + c16 * 8;
            __bf16* la = &sA[buf][row * LDSROW + c16 * 8];
            __bf16* lb = &sB[buf][row * LDSROW + c16 * 8];
#if HAVE_ASYNC_LDS
            async_cp16(ga, la);
            async_cp16(gb, lb);
#else
            *(v8bf*)la = *(const v8bf*)ga;
            *(v8bf*)lb = *(const v8bf*)gb;
#endif
        }
    };

    v8f acc[4][4] = {};
    const int T = K / 32;
    issue(0, 0);

    for (int t = 0; t < T; ++t) {
        if (t + 1 < T) {
            issue((t + 1) * 32, (t + 1) & 1);
            wait_async<8>();          // previous stage done; only t+1 in flight
        } else {
            wait_async<0>();
        }
        __syncthreads();              // all waves' LDS writes for buf[t&1] visible

        const __bf16* pa = &sA[t & 1][0];
        const __bf16* pb = &sB[t & 1][0];
        v16bf af[4], bfr[4];
#pragma unroll
        for (int mi = 0; mi < 4; ++mi) {
            const __bf16* p = pa + (wm + 16 * mi + lr) * LDSROW + 8 * g;
            v8bf lo = *(const v8bf*)(p);
            v8bf hi = *(const v8bf*)(p + 16);
#pragma unroll
            for (int e = 0; e < 8; ++e) { af[mi][e] = lo[e]; af[mi][8 + e] = hi[e]; }
        }
#pragma unroll
        for (int ni = 0; ni < 4; ++ni) {
            const __bf16* p = pb + (wn + 16 * ni + lr) * LDSROW + 8 * g;
            v8bf lo = *(const v8bf*)(p);
            v8bf hi = *(const v8bf*)(p + 16);
#pragma unroll
            for (int e = 0; e < 8; ++e) { bfr[ni][e] = lo[e]; bfr[ni][8 + e] = hi[e]; }
        }
#pragma unroll
        for (int mi = 0; mi < 4; ++mi)
#pragma unroll
            for (int ni = 0; ni < 4; ++ni)
                acc[mi][ni] = __builtin_amdgcn_wmma_f32_16x16x32_bf16(
                    false, af[mi], false, bfr[ni], (short)0, acc[mi][ni], false, false);
        __syncthreads();              // protect buf[t&1] before iter t+1 overwrites it
    }

    epi_store<FLAGS>(OUT, ldo, bm + wm, bn + wn, acc, lane, bias_m, bias_n, nullptr);
}

// kv split-K partials: grid = (bh=64, part=16); OUT[e][d] f32 per (part,bh).
__global__ __launch_bounds__(32)
void k_kv_partial(const __bf16* __restrict__ kT, const __bf16* __restrict__ vT,
                  float* __restrict__ kvp)
{
    const int bh = blockIdx.x, part = blockIdx.y;
    const int b = bh >> 4, h = bh & 15;
    const size_t offs = (size_t)(h * HDIM) * TOKENS + (size_t)b * SEQ + (size_t)part * (SEQ / KVPARTS);
    float* out = kvp + ((size_t)part * (BATCH * HEADS) + bh) * (HDIM * HDIM);
    wave_gemm_64x64<FLAG_OUTF32>(kT + offs, TOKENS, vT + offs, TOKENS,
                                 out, HDIM, 0, 0, SEQ / KVPARTS,
                                 nullptr, nullptr, nullptr);
}

__global__ void k_kv_reduce(const float* __restrict__ kvp, __bf16* __restrict__ kvT)
{
    const int bh = blockIdx.x;
    for (int i = threadIdx.x; i < HDIM * HDIM; i += 256) {
        float s = 0.0f;
        for (int p = 0; p < KVPARTS; ++p)
            s += kvp[((size_t)p * (BATCH * HEADS) + bh) * (HDIM * HDIM) + i];
        kvT[(size_t)bh * (HDIM * HDIM) + i] = (__bf16)s;
    }
}

__global__ void k_ksum(const __bf16* __restrict__ kT, float* __restrict__ ksum)
{
    __shared__ float red[128];
    const int b = blockIdx.x >> 10;
    const int c = blockIdx.x & 1023;
    const __bf16* row = kT + (size_t)c * TOKENS + (size_t)b * SEQ;
    float s = 0.0f;
    for (int i = threadIdx.x; i < SEQ; i += 128) s += (float)row[i];
    red[threadIdx.x] = s;
    __syncthreads();
    for (int st = 64; st > 0; st >>= 1) {
        if (threadIdx.x < st) red[threadIdx.x] += red[threadIdx.x + st];
        __syncthreads();
    }
    if (threadIdx.x == 0) ksum[(size_t)b * DIM + c] = red[0];
}

__global__ void k_norm(const __bf16* __restrict__ qb, const float* __restrict__ ksum,
                       float* __restrict__ norm)
{
    const int idx = blockIdx.x * 256 + threadIdx.x;
    const int s = idx & (SEQ - 1);
    const int h = (idx >> 12) & (HEADS - 1);
    const int b = idx >> 16;
    const __bf16* qr = qb + ((size_t)b * SEQ + s) * DIM + h * HDIM;
    const float*  ks = ksum + (size_t)b * DIM + h * HDIM;
    float acc = 0.0f;
#pragma unroll 8
    for (int d = 0; d < HDIM; ++d) acc += (float)qr[d] * ks[d];
    norm[((size_t)b * HEADS + h) * SEQ + s] = acc + EPSV;
}

__global__ __launch_bounds__(32)
void k_attn(const __bf16* __restrict__ kvT, const __bf16* __restrict__ qb,
            const float* __restrict__ norm, __bf16* __restrict__ attnb)
{
    const int bh = blockIdx.x, tn = blockIdx.y;
    const int b = bh >> 4, h = bh & 15;
    const __bf16* A  = kvT + (size_t)bh * (HDIM * HDIM);
    const __bf16* Bp = qb + (size_t)b * SEQ * DIM + h * HDIM;
    __bf16* out = attnb + (size_t)b * SEQ * DIM + h * HDIM;
    const float* nrm = norm + (size_t)bh * SEQ;
    wave_gemm_64x64<FLAG_NORM>(A, HDIM, Bp, DIM, out, DIM,
                               0, tn * 64, HDIM, nullptr, nullptr, nrm);
}

__global__ void k_cvt(const float* __restrict__ s, __bf16* __restrict__ d, size_t n)
{
    size_t i = ((size_t)blockIdx.x * blockDim.x + threadIdx.x) * 8;
    if (i + 8 <= n) {
        v4f a = *(const v4f*)(s + i);
        v4f b = *(const v4f*)(s + i + 4);
        v8bf o;
#pragma unroll
        for (int j = 0; j < 4; ++j) { o[j] = (__bf16)a[j]; o[4 + j] = (__bf16)b[j]; }
        *(v8bf*)(d + i) = o;
    }
}

extern "C" void kernel_launch(void* const* d_in, const int* in_sizes, int n_in,
                              void* d_out, int out_size, void* d_ws, size_t ws_size,
                              hipStream_t stream)
{
    (void)in_sizes; (void)n_in; (void)out_size; (void)ws_size;
    const float* x  = (const float*)d_in[0];
    const float* Wq = (const float*)d_in[1];
    const float* bq = (const float*)d_in[2];
    const float* Wk = (const float*)d_in[3];
    const float* bk = (const float*)d_in[4];
    const float* Wv = (const float*)d_in[5];
    const float* bv = (const float*)d_in[6];
    const float* Wo = (const float*)d_in[7];
    const float* bo = (const float*)d_in[8];
    float* out = (float*)d_out;

    char* w = (char*)d_ws;
    size_t off = 0;
    auto take = [&](size_t bytes) -> char* {
        char* p = w + off;
        off += (bytes + 255) & ~(size_t)255;
        return p;
    };

    __bf16* Xb   = (__bf16*)take((size_t)TOKENS * DIM * 2);
    __bf16* Wqb  = (__bf16*)take((size_t)DIM * DIM * 2);
    __bf16* Wkb  = (__bf16*)take((size_t)DIM * DIM * 2);
    __bf16* Wvb  = (__bf16*)take((size_t)DIM * DIM * 2);
    __bf16* Wob  = (__bf16*)take((size_t)DIM * DIM * 2);
    __bf16* qb   = (__bf16*)take((size_t)TOKENS * DIM * 2);
    __bf16* kTb  = (__bf16*)take((size_t)DIM * TOKENS * 2);
    __bf16* vTb  = (__bf16*)take((size_t)DIM * TOKENS * 2);
    float*  kvp  = (float*)take((size_t)KVPARTS * BATCH * HEADS * HDIM * HDIM * 4);
    __bf16* kvTb = (__bf16*)take((size_t)BATCH * HEADS * HDIM * HDIM * 2);
    float*  ksum = (float*)take((size_t)BATCH * DIM * 4);
    float*  norm = (float*)take((size_t)BATCH * HEADS * SEQ * 4);
    __bf16* attnb = Xb;   // X dead after vT projection; reuse its 32 MB

    // 1) f32 -> bf16 converts
    k_cvt<<<dim3((TOKENS * DIM) / 8 / 256), 256, 0, stream>>>(x, Xb, (size_t)TOKENS * DIM);
    k_cvt<<<dim3((DIM * DIM) / 8 / 256), 256, 0, stream>>>(Wq, Wqb, (size_t)DIM * DIM);
    k_cvt<<<dim3((DIM * DIM) / 8 / 256), 256, 0, stream>>>(Wk, Wkb, (size_t)DIM * DIM);
    k_cvt<<<dim3((DIM * DIM) / 8 / 256), 256, 0, stream>>>(Wv, Wvb, (size_t)DIM * DIM);
    k_cvt<<<dim3((DIM * DIM) / 8 / 256), 256, 0, stream>>>(Wo, Wob, (size_t)DIM * DIM);

    // 2) q[s][c] = elu(X Wq^T + bq)+1   (A=Wq: m=c, B=X: n=s)
    k_gemm_lds<FLAG_ELU | FLAG_BIAS_M><<<dim3(DIM / 128, TOKENS / 128), 128, 0, stream>>>(
        Wqb, DIM, Xb, DIM, qb, DIM, DIM, bq, nullptr);

    // 3) kT[c][s], vT[c][s]   (A=X: m=s, B=W: n=c) -> transposed layouts
    k_gemm_lds<FLAG_ELU | FLAG_BIAS_N><<<dim3(TOKENS / 128, DIM / 128), 128, 0, stream>>>(
        Xb, DIM, Wkb, DIM, kTb, TOKENS, DIM, nullptr, bk);
    k_gemm_lds<FLAG_BIAS_N><<<dim3(TOKENS / 128, DIM / 128), 128, 0, stream>>>(
        Xb, DIM, Wvb, DIM, vTb, TOKENS, DIM, nullptr, bv);

    // 4) kv aggregation (split-K) + reduce -> kvT[bh][e][d]
    k_kv_partial<<<dim3(BATCH * HEADS, KVPARTS), 32, 0, stream>>>(kTb, vTb, kvp);
    k_kv_reduce<<<dim3(BATCH * HEADS), 256, 0, stream>>>(kvp, kvTb);

    // 5) k_sum and per-token normalizer (includes +eps)
    k_ksum<<<dim3(BATCH * DIM), 128, 0, stream>>>(kTb, ksum);
    k_norm<<<dim3((BATCH * HEADS * SEQ) / 256), 256, 0, stream>>>(qb, ksum, norm);

    // 6) attn[s][c] = (q @ kv) / normalizer
    k_attn<<<dim3(BATCH * HEADS, SEQ / 64), 32, 0, stream>>>(kvTb, qb, norm, attnb);

    // 7) out[s][c_out] = attn Wo^T + bo (f32)
    k_gemm_lds<FLAG_BIAS_M | FLAG_OUTF32><<<dim3(DIM / 128, TOKENS / 128), 128, 0, stream>>>(
        Wob, DIM, attnb, DIM, out, DIM, DIM, bo, nullptr);
}
